// MulDecoder_74560632258948
// MI455X (gfx1250) — compile-verified, matrix operand order checked by reading, not run on hardware
//
#include <hip/hip_runtime.h>

// ---------------- problem constants (from reference) ----------------
#define B_   4
#define T_   128
#define S_   256
#define H_   512
#define NH_  8
#define DK_  64
#define F_   2048
#define E_   32

typedef __attribute__((ext_vector_type(16))) _Float16 v16h;
typedef __attribute__((ext_vector_type(8)))  float    v8f;

#define FLAG_TRANSB 1
#define FLAG_RELU   2
#define FLAG_SPLIT  4

// ---------------- WMMA fragment loaders (CDNA5 16x16x32 f16 layout) ---------
// A 16x32: lane holds row m=lane&15; halves j<4 -> K=koff+2j+{0,1},
//          j>=4 -> K=koff+2j+8+{0,1}, koff=8*(lane>>4).
// B 32x16 mirrors A with col n=lane&15 (rows striped across lanes per VGPR).
__device__ __forceinline__ v16h frag_row(const float* __restrict__ p) {
  v16h r;
#pragma unroll
  for (int j = 0; j < 8; ++j) {
    int kk = (j < 4) ? 2 * j : 2 * j + 8;
    float2 f = *(const float2*)(p + kk);
    r[2 * j]     = (_Float16)f.x;
    r[2 * j + 1] = (_Float16)f.y;
  }
  return r;
}

__device__ __forceinline__ v16h frag_col(const float* __restrict__ p, int ld) {
  v16h r;
#pragma unroll
  for (int j = 0; j < 8; ++j) {
    int kk = (j < 4) ? 2 * j : 2 * j + 8;
    r[2 * j]     = (_Float16)p[(long)kk * ld];
    r[2 * j + 1] = (_Float16)p[(long)(kk + 1) * ld];
  }
  return r;
}

// ---------------- generic batched WMMA GEMM -------------------------------
// C[z] = act(alpha * A[z] @ B[z] + bias) (+ add)
// A: [M,K] row-major (lda), per-batch stride sA
// B: normal [K,N] (ldb) or FLAG_TRANSB: [N,K] row-major (ldb)
// C offset per batch: (z/cMod)*sC0 + (z%cMod)*sC1
// FLAG_SPLIT: col n -> (h=n>>6, d=n&63), element at h*(M*64) + m*64 + d
// Block: 4 waves tiled 2x2 over (M,N); block tile 32 rows x 128 cols.
// Each wave: 16 rows x 64 cols = 4 accumulators; A fragment loaded once per
// 32-K step and reused across 4 back-to-back WMMAs.
// CONTRACT: M % 32 == 0, N % 64 == 0, K % 32 == 0 (all shapes in this model
// satisfy this), so active waves always own 4 full n-tiles -> straight-line
// inner loop, EXEC all-ones around every WMMA.
__global__ __launch_bounds__(128)
void k_gemm(const float* __restrict__ A, long sA, int lda,
            const float* __restrict__ Bm, long sB, int ldb,
            const float* __restrict__ bias,
            const float* __restrict__ add, long sAd0, long sAd1,
            float* __restrict__ C, long sC0, long sC1, int cMod, int ldc,
            int M, int N, int K, float alpha, int flags) {
  const int lane = threadIdx.x;
  const int wv   = threadIdx.y;
  const int m0    = blockIdx.y * 32 + (wv & 1) * 16;
  const int nbase = blockIdx.x * 128 + (wv >> 1) * 64;
  if (nbase >= N) return;                 // wave-uniform: EXEC stays all-ones
  const int z = blockIdx.z;
  const float* Ab = A + (long)z * sA;
  const float* Bb = Bm + (long)z * sB;
  const int r    = lane & 15;
  const int koff = (lane >> 4) << 3;
  const float* arow = Ab + (long)(m0 + r) * lda + koff;

  v8f acc[4] = {v8f{}, v8f{}, v8f{}, v8f{}};
  if (flags & FLAG_TRANSB) {
    const float* brow = Bb + (long)(nbase + r) * ldb + koff;
    for (int k0 = 0; k0 < K; k0 += 32) {
      if (k0 + 64 <= K) __builtin_prefetch(arow + k0 + 32, 0, 0);
      v16h a = frag_row(arow + k0);
#pragma unroll
      for (int nt = 0; nt < 4; ++nt) {
        v16h b = frag_row(brow + (long)(16 * nt) * ldb + k0);
        acc[nt] = __builtin_amdgcn_wmma_f32_16x16x32_f16(
            false, a, false, b, (short)0, acc[nt], false, false);
      }
    }
  } else {
    for (int k0 = 0; k0 < K; k0 += 32) {
      if (k0 + 64 <= K) __builtin_prefetch(arow + k0 + 32, 0, 0);
      v16h a = frag_row(arow + k0);
      const float* brow = Bb + (long)(k0 + koff) * ldb + nbase + r;
#pragma unroll
      for (int nt = 0; nt < 4; ++nt) {
        v16h b = frag_col(brow + 16 * nt, ldb);
        acc[nt] = __builtin_amdgcn_wmma_f32_16x16x32_f16(
            false, a, false, b, (short)0, acc[nt], false, false);
      }
    }
  }

  // D layout: VGPR i -> row m0 + i + 8*(lane>>4), col ntile + (lane&15)
  const int mhi = (lane >> 4) << 3;
  const long cz = (long)(z / cMod) * sC0 + (long)(z % cMod) * sC1;
  const long az = (long)(z / cMod) * sAd0 + (long)(z % cMod) * sAd1;
#pragma unroll
  for (int nt = 0; nt < 4; ++nt) {
    const int n = nbase + 16 * nt + (lane & 15);
#pragma unroll
    for (int i = 0; i < 8; ++i) {
      const int m = m0 + mhi + i;
      float v = acc[nt][i] * alpha;
      if (bias) v += bias[n];
      if (flags & FLAG_RELU) v = v > 0.f ? v : 0.f;
      if (flags & FLAG_SPLIT) {
        const int h = n >> 6, d = n & 63;
        C[cz + (long)h * (long)M * DK_ + (long)m * DK_ + d] = v;
      } else {
        if (add) v += add[az + (long)m * ldc + n];
        C[cz + (long)m * ldc + n] = v;
      }
    }
  }
}

// ---------------- layer norm (one 512-wide row per 256-thread block) -------
__global__ __launch_bounds__(256)
void k_ln(const float* __restrict__ x, const float* __restrict__ g,
          const float* __restrict__ b, float* __restrict__ y) {
  __shared__ float red[256];
  const int row = blockIdx.x, tid = threadIdx.x;
  const float* xr = x + (long)row * H_;
  const float x0 = xr[tid], x1 = xr[tid + 256];
  red[tid] = x0 + x1;
  __syncthreads();
  for (int st = 128; st > 0; st >>= 1) {
    if (tid < st) red[tid] += red[tid + st];
    __syncthreads();
  }
  const float mu = red[0] * (1.f / H_);
  __syncthreads();
  const float d0 = x0 - mu, d1 = x1 - mu;
  red[tid] = d0 * d0 + d1 * d1;
  __syncthreads();
  for (int st = 128; st > 0; st >>= 1) {
    if (tid < st) red[tid] += red[tid + st];
    __syncthreads();
  }
  const float rs = rsqrtf(red[0] * (1.f / H_) + 1e-6f);
  float* yr = y + (long)row * H_;
  yr[tid]       = d0 * rs * g[tid] + b[tid];
  yr[tid + 256] = d1 * rs * g[tid + 256] + b[tid + 256];
}

// ---------------- masked softmax over key dim (row per block) --------------
// row = ((b*NH + h)*T + t); self: mask[b*Sk+s] || s>t ; cross: mask[b*Sk+s]
__global__ __launch_bounds__(256)
void k_softmax(float* __restrict__ w, const int* __restrict__ mask,
               int Sk, int selfm) {
  __shared__ float red[256];
  const int row = blockIdx.x, tid = threadIdx.x;
  const int t = row % T_;
  const int b = row / (NH_ * T_);
  float* p = w + (long)row * Sk;
  const bool active = tid < Sk;
  float v = -3.0e38f;
  if (active) {
    const bool mk = (mask[b * Sk + tid] != 0) || (selfm && tid > t);
    v = mk ? -1e18f : p[tid];
  }
  red[tid] = v;
  __syncthreads();
  for (int st = 128; st > 0; st >>= 1) {
    if (tid < st) red[tid] = fmaxf(red[tid], red[tid + st]);
    __syncthreads();
  }
  const float mx = red[0];
  __syncthreads();
  const float e = active ? __expf(v - mx) : 0.f;
  red[tid] = e;
  __syncthreads();
  for (int st = 128; st > 0; st >>= 1) {
    if (tid < st) red[tid] += red[tid + st];
    __syncthreads();
  }
  const float inv = 1.f / red[0];
  if (active) p[tid] = e * inv;
}

// ---------------- small elementwise kernels --------------------------------
__global__ void k_zero(float* p, int n) {
  int i = blockIdx.x * blockDim.x + threadIdx.x;
  if (i < n) p[i] = 0.f;
}
__global__ void k_add(const float* __restrict__ a, const float* __restrict__ b,
                      float* __restrict__ c, int n) {
  int i = blockIdx.x * blockDim.x + threadIdx.x;
  if (i < n) c[i] = a[i] + b[i];
}
__global__ void k_addtime(float* __restrict__ x, const float* __restrict__ tim,
                          int n) {
  int i = blockIdx.x * blockDim.x + threadIdx.x;
  if (i < n) {
    int col = i % H_;
    int t = (i / H_) % T_;
    x[i] += tim[t * H_ + col];
  }
}
__global__ void k_expacc(float* __restrict__ acc, const float* __restrict__ t,
                         const float* __restrict__ ae, int e, int n) {
  int i = blockIdx.x * blockDim.x + threadIdx.x;
  if (i < n) {
    int b = i / (T_ * H_);
    acc[i] += ae[b * E_ + e] * t[i];
  }
}
__global__ void k_meanheads(const float* __restrict__ w, float* __restrict__ o,
                            int n) {
  int i = blockIdx.x * blockDim.x + threadIdx.x;
  if (i < n) {
    int s = i % S_;
    int t = (i / S_) % T_;
    int b = i / (S_ * T_);
    float acc = 0.f;
#pragma unroll
    for (int h = 0; h < NH_; ++h)
      acc += w[((long)(b * NH_ + h) * T_ + t) * S_ + s];
    o[i] = acc * (1.f / NH_);
  }
}

// ---------------- host-side orchestration ----------------------------------
static inline void gemm(hipStream_t st, const float* A, long sA, int lda,
                        const float* Bm, long sB, int ldb, const float* bias,
                        const float* add, long sAd0, long sAd1,
                        float* C, long sC0, long sC1, int cMod, int ldc,
                        int M, int N, int K, float alpha, int flags,
                        int batch) {
  dim3 g((N + 127) / 128, (M + 31) / 32, batch), b(32, 4, 1);
  k_gemm<<<g, b, 0, st>>>(A, sA, lda, Bm, sB, ldb, bias, add, sAd0, sAd1, C,
                          sC0, sC1, cMod, ldc, M, N, K, alpha, flags);
}

// Param leaves in jax alphabetical order within each dict:
// W1 W2 b1 b2 cWk cWo cWq cWv ln1_b ln1_g ln2_b ln2_g ln3_b ln3_g sWk sWo sWq sWv
struct LayerP {
  const float *W1, *W2, *b1, *b2, *cWk, *cWo, *cWq, *cWv;
  const float *ln1_b, *ln1_g, *ln2_b, *ln2_g, *ln3_b, *ln3_g;
  const float *sWk, *sWo, *sWq, *sWv;
};

static LayerP get_layer(void* const* d_in, int base, long idx) {
  const long HH = (long)H_ * H_, HF = (long)H_ * F_;
  auto g = [&](int i, long per) { return (const float*)d_in[base + i] + idx * per; };
  LayerP p;
  p.W1 = g(0, HF);   p.W2 = g(1, HF);   p.b1 = g(2, F_);   p.b2 = g(3, H_);
  p.cWk = g(4, HH);  p.cWo = g(5, HH);  p.cWq = g(6, HH);  p.cWv = g(7, HH);
  p.ln1_b = g(8, H_);  p.ln1_g = g(9, H_);
  p.ln2_b = g(10, H_); p.ln2_g = g(11, H_);
  p.ln3_b = g(12, H_); p.ln3_g = g(13, H_);
  p.sWk = g(14, HH); p.sWo = g(15, HH); p.sWq = g(16, HH); p.sWv = g(17, HH);
  return p;
}

struct WS {
  float *xn, *q, *kb, *vb, *attw, *attno, *res1, *ffh;
};

static void run_layer(hipStream_t st, const LayerP& p, const float* xin,
                      float* xout, const float* enc, const int* msrc,
                      const int* mtrg, float* attn_dist, const WS& w) {
  const float qs = 0.125f;  // dk^-0.5
  const int NROW = B_ * T_;
  // ---- self attention ----
  k_ln<<<NROW, 256, 0, st>>>(xin, p.ln1_g, p.ln1_b, w.xn);
  gemm(st, w.xn, (long)T_ * H_, H_, p.sWq, 0, H_, nullptr, nullptr, 0, 0,
       w.q, (long)NH_ * T_ * DK_, 0, 1, 0, T_, H_, H_, qs, FLAG_SPLIT, B_);
  gemm(st, w.xn, (long)T_ * H_, H_, p.sWk, 0, H_, nullptr, nullptr, 0, 0,
       w.kb, (long)NH_ * T_ * DK_, 0, 1, 0, T_, H_, H_, 1.f, FLAG_SPLIT, B_);
  gemm(st, w.xn, (long)T_ * H_, H_, p.sWv, 0, H_, nullptr, nullptr, 0, 0,
       w.vb, (long)NH_ * T_ * DK_, 0, 1, 0, T_, H_, H_, 1.f, FLAG_SPLIT, B_);
  gemm(st, w.q, (long)T_ * DK_, DK_, w.kb, (long)T_ * DK_, DK_, nullptr,
       nullptr, 0, 0, w.attw, (long)T_ * T_, 0, 1, T_, T_, T_, DK_, 1.f,
       FLAG_TRANSB, B_ * NH_);
  k_softmax<<<B_ * NH_ * T_, 256, 0, st>>>(w.attw, mtrg, T_, 1);
  gemm(st, w.attw, (long)T_ * T_, T_, w.vb, (long)T_ * DK_, DK_, nullptr,
       nullptr, 0, 0, w.attno, (long)T_ * H_, (long)DK_, NH_, H_, T_, DK_, T_,
       1.f, 0, B_ * NH_);
  gemm(st, w.attno, 0, H_, p.sWo, 0, H_, nullptr, xin, 0, 0,
       w.res1, 0, 0, 1, H_, NROW, H_, H_, 1.f, 0, 1);
  // ---- cross attention ----
  k_ln<<<NROW, 256, 0, st>>>(w.res1, p.ln2_g, p.ln2_b, w.xn);
  gemm(st, w.xn, (long)T_ * H_, H_, p.cWq, 0, H_, nullptr, nullptr, 0, 0,
       w.q, (long)NH_ * T_ * DK_, 0, 1, 0, T_, H_, H_, qs, FLAG_SPLIT, B_);
  gemm(st, enc, (long)S_ * H_, H_, p.cWk, 0, H_, nullptr, nullptr, 0, 0,
       w.kb, (long)NH_ * S_ * DK_, 0, 1, 0, S_, H_, H_, 1.f, FLAG_SPLIT, B_);
  gemm(st, enc, (long)S_ * H_, H_, p.cWv, 0, H_, nullptr, nullptr, 0, 0,
       w.vb, (long)NH_ * S_ * DK_, 0, 1, 0, S_, H_, H_, 1.f, FLAG_SPLIT, B_);
  gemm(st, w.q, (long)T_ * DK_, DK_, w.kb, (long)S_ * DK_, DK_, nullptr,
       nullptr, 0, 0, w.attw, (long)T_ * S_, 0, 1, S_, T_, S_, DK_, 1.f,
       FLAG_TRANSB, B_ * NH_);
  k_softmax<<<B_ * NH_ * T_, 256, 0, st>>>(w.attw, msrc, S_, 0);
  if (attn_dist)
    k_meanheads<<<(B_ * T_ * S_ + 255) / 256, 256, 0, st>>>(w.attw, attn_dist,
                                                            B_ * T_ * S_);
  gemm(st, w.attw, (long)T_ * S_, S_, w.vb, (long)S_ * DK_, DK_, nullptr,
       nullptr, 0, 0, w.attno, (long)T_ * H_, (long)DK_, NH_, H_, T_, DK_, S_,
       1.f, 0, B_ * NH_);
  gemm(st, w.attno, 0, H_, p.cWo, 0, H_, nullptr, w.res1, 0, 0,
       xout, 0, 0, 1, H_, NROW, H_, H_, 1.f, 0, 1);
  // ---- FFN ----
  k_ln<<<NROW, 256, 0, st>>>(xout, p.ln3_g, p.ln3_b, w.xn);
  gemm(st, w.xn, 0, H_, p.W1, 0, F_, p.b1, nullptr, 0, 0,
       w.ffh, 0, 0, 1, F_, NROW, F_, H_, 1.f, FLAG_RELU, 1);
  gemm(st, w.ffh, 0, F_, p.W2, 0, H_, p.b2, xout, 0, 0,
       xout, 0, 0, 1, H_, NROW, H_, F_, 1.f, 0, 1);
}

extern "C" void kernel_launch(void* const* d_in, const int* in_sizes, int n_in,
                              void* d_out, int out_size, void* d_ws,
                              size_t ws_size, hipStream_t stream) {
  (void)in_sizes; (void)n_in; (void)out_size; (void)ws_size;
  // Input order: top-level dict insertion order; nested param dicts flattened
  // alphabetically (jax tree order); stacked expert/dec leaves lead with E/L.
  const float* inputs = (const float*)d_in[0];
  const float* enc    = (const float*)d_in[1];
  const int*   msrc   = (const int*)d_in[2];
  const int*   mtrg   = (const int*)d_in[3];
  const float* ae     = (const float*)d_in[4];
  const float* timing = (const float*)d_in[5];
  const float* projW  = (const float*)d_in[6];
  const float* lng    = (const float*)d_in[61];
  const float* lnb    = (const float*)d_in[62];

  float* wsf = (float*)d_ws;
  size_t off = 0;
  auto alloc = [&](size_t n) { float* p = wsf + off; off += n; return p; };
  const size_t NX = (size_t)B_ * T_ * H_;

  float* xw     = alloc(NX);
  float* basic  = alloc(NX);
  float* expacc = alloc(NX);
  float* tmpo   = alloc(NX);
  float* curA   = alloc(NX);
  float* curB   = alloc(NX);
  WS w;
  w.xn    = alloc(NX);
  w.q     = alloc(NX);
  w.res1  = alloc(NX);
  w.attno = alloc(NX);
  w.kb    = alloc((size_t)B_ * NH_ * S_ * DK_);
  w.vb    = alloc((size_t)B_ * NH_ * S_ * DK_);
  w.attw  = alloc((size_t)B_ * NH_ * T_ * S_);
  w.ffh   = alloc((size_t)B_ * T_ * F_);

  const int nthr = 256;
  // x = inputs @ proj_W + timing_signal[:T]
  gemm(stream, inputs, 0, H_, projW, 0, H_, nullptr, nullptr, 0, 0,
       xw, 0, 0, 1, H_, B_ * T_, H_, H_, 1.f, 0, 1);
  k_addtime<<<((int)NX + nthr - 1) / nthr, nthr, 0, stream>>>(xw, timing,
                                                              (int)NX);
  // basic layer
  LayerP bp = get_layer(d_in, 7, 0);
  run_layer(stream, bp, xw, basic, enc, msrc, mtrg, nullptr, w);
  // expert layers, weighted accumulation
  k_zero<<<((int)NX + nthr - 1) / nthr, nthr, 0, stream>>>(expacc, (int)NX);
  for (int e = 0; e < E_; ++e) {
    LayerP ep = get_layer(d_in, 25, e);
    run_layer(stream, ep, xw, tmpo, enc, msrc, mtrg, nullptr, w);
    k_expacc<<<((int)NX + nthr - 1) / nthr, nthr, 0, stream>>>(expacc, tmpo,
                                                               ae, e, (int)NX);
  }
  k_add<<<((int)NX + nthr - 1) / nthr, nthr, 0, stream>>>(expacc, basic, curA,
                                                          (int)NX);
  // final decoder layers
  float* yout = (float*)d_out;
  float* attn_dist = yout + NX;
  LayerP d0 = get_layer(d_in, 43, 0);
  run_layer(stream, d0, curA, curB, enc, msrc, mtrg, nullptr, w);
  LayerP d1 = get_layer(d_in, 43, 1);
  run_layer(stream, d1, curB, curA, enc, msrc, mtrg, attn_dist, w);
  // final LN -> y
  k_ln<<<B_ * T_, 256, 0, stream>>>(curA, lng, lnb, yout);
}
